// TemporalTransformerConv_61607010894577
// MI455X (gfx1250) — matrix operand order repaired and checked by reading, not compiled, not run on hardware
//
#include <hip/hip_runtime.h>

// ---------------- problem constants ----------------
#define ND 50000      // n_dst
#define NS 200000     // n_src
#define EE 250000     // edges
#define DNODE 128
#define DEDGE 64
#define DTIME 64
#define DOUT 128
#define DQ 192        // dim_node + dim_time
#define DK 256        // dim_node + dim_edge + dim_time
#define NHEAD 2
#define HDIM 64
#define LN_EPS 1e-5f

typedef __attribute__((ext_vector_type(16))) __bf16 v16bf;
typedef __attribute__((ext_vector_type(8)))  __bf16 v8bf;
typedef __attribute__((ext_vector_type(8)))  float  v8f;

// ---------------- async (CDNA5 GLOBAL_*_ASYNC_*_LDS) with safe fallback ----------------
#if defined(__has_builtin)
#if __has_builtin(__builtin_amdgcn_global_load_async_to_lds_b128)
#define HAVE_ASYNC 1
#endif
#endif
#ifndef HAVE_ASYNC
#define HAVE_ASYNC 0
#endif

#if HAVE_ASYNC
// builtin expects pointers to int-vector(16B); AS1 = global, AS3 = LDS
typedef int v4i __attribute__((vector_size(16)));
typedef __attribute__((address_space(1))) v4i gas_v4i;
typedef __attribute__((address_space(3))) v4i las_v4i;
__device__ __forceinline__ void async_ld16(const float* g, float* l) {
  __builtin_amdgcn_global_load_async_to_lds_b128((gas_v4i*)g, (las_v4i*)l, 0, 0);
}
__device__ __forceinline__ void async_wait0() {
#if __has_builtin(__builtin_amdgcn_s_wait_asynccnt)
  __builtin_amdgcn_s_wait_asynccnt(0);
#else
  asm volatile("s_wait_asynccnt 0x0" ::: "memory");
#endif
}
__device__ __forceinline__ void async_st16(float* g, const float* l) {
#if __has_builtin(__builtin_amdgcn_global_store_async_from_lds_b128)
  __builtin_amdgcn_global_store_async_from_lds_b128((gas_v4i*)g, (las_v4i*)l, 0, 0);
#else
  *(float4*)g = *(const float4*)l;
#endif
}
#else
__device__ __forceinline__ void async_ld16(const float* g, float* l) {
  *(float4*)l = *(const float4*)g;
}
__device__ __forceinline__ void async_wait0() {}
__device__ __forceinline__ void async_st16(float* g, const float* l) {
  *(float4*)g = *(const float4*)l;
}
#endif

// ---------------- helpers ----------------
__device__ __forceinline__ __bf16 f2bf(float f) {
  unsigned u = __float_as_uint(f);
  unsigned r = u + 0x7FFFu + ((u >> 16) & 1u);   // round-to-nearest-even
  unsigned short s = (unsigned short)(r >> 16);
  return __builtin_bit_cast(__bf16, s);
}

// order-preserving float<->uint encoding for atomic segment-max
__device__ __forceinline__ unsigned fenc(float f) {
  unsigned u = __float_as_uint(f);
  return (u & 0x80000000u) ? ~u : (u | 0x80000000u);
}
__device__ __forceinline__ float fdec(unsigned u) {
  unsigned v = (u & 0x80000000u) ? (u & 0x7FFFFFFFu) : ~u;
  return __uint_as_float(v);
}
#define ENC_NEG_INF 0x007FFFFFu   // fenc(-inf)

__device__ __forceinline__ v8f wmma_bf16(v16bf a, v16bf b, v8f c) {
  return __builtin_amdgcn_wmma_f32_16x16x32_bf16(
      false, a, false, b, (short)0, c, false, false);
}

// A-fragment (16x32 bf16, ISA 7.12.2)
__device__ __forceinline__ v16bf load_a_frag(const __bf16* base, int stride, int kt, int lane) {
  int m = lane & 15, kh = (lane >> 4) & 1;
  const __bf16* p = base + m * stride + kt * 32 + kh * 8;
  v8bf lo = *(const v8bf*)p;
  v8bf hi = *(const v8bf*)(p + 16);
  v16bf r;
#pragma unroll
  for (int i = 0; i < 8; ++i) { r[i] = lo[i]; r[i + 8] = hi[i]; }
  return r;
}

// B-fragments pre-swizzled: ((kt*8+nt)*32 + lane)*16 contiguous bf16
__device__ __forceinline__ v16bf load_b_frag(const __bf16* w, int kt, int nt, int lane) {
  return *(const v16bf*)(w + (((kt * 8 + nt) * 32) + lane) * 16);
}

// ---------------- kernel 0: init accumulators ----------------
__global__ void k_init(float* agg, float* den, unsigned* mEnc) {
  int i = blockIdx.x * blockDim.x + threadIdx.x;
  if (i < ND * DOUT) agg[i] = 0.0f;
  if (i < ND * NHEAD) { den[i] = 0.0f; mEnc[i] = ENC_NEG_INF; }
}

// ---------------- kernel 1: weight prep (fp32 -> bf16, fragment order) ----------------
__global__ void k_prep(const float* __restrict__ W, __bf16* __restrict__ dst, int K) {
  int idx = blockIdx.x * blockDim.x + threadIdx.x;
  if (idx >= K * 128) return;
  int e = idx & 15;
  int lane = (idx >> 4) & 31;
  int nt = (idx >> 9) & 7;
  int kt = idx >> 12;
  int k = kt * 32 + (lane >> 4) * 8 + (e & 7) + (e >> 3) * 16;
  int n = nt * 16 + (lane & 15);
  dst[idx] = f2bf(W[k * 128 + n]);
}

// ---------------- kernel 2: Q projection GEMM (ND x 192 x 128) ----------------
#define QWPB 4
__global__ __launch_bounds__(32 * QWPB) void k_q(
    const float* __restrict__ h, const float* __restrict__ time_b,
    const __bf16* __restrict__ wqb, const float* __restrict__ wq_b,
    float* __restrict__ Q) {
  __shared__ __bf16 sA[QWPB][16 * DQ];      // 6 KB / wave
  __shared__ float  sS[QWPB][16 * DOUT];    // 8 KB / wave: staging h f32, later Q f32
  int wave = threadIdx.x >> 5, lane = threadIdx.x & 31;
  int tile = blockIdx.x * QWPB + wave;
  __bf16* A = sA[wave];
  float*  S = sS[wave];

  // coalesced async stage: one row (512B) per instruction
#pragma unroll 4
  for (int r = 0; r < 16; ++r) {
    int rr = tile * 16 + r;
    int rc = (rr < ND) ? rr : (ND - 1);
    async_ld16(h + (size_t)rc * DNODE + lane * 4, S + r * DOUT + lane * 4);
  }
  // time columns (same for all rows) while loads are in flight
  for (int c = lane; c < 64; c += 32) {
    __bf16 tv = f2bf(__cosf(time_b[c]));
#pragma unroll
    for (int r = 0; r < 16; ++r) A[r * DQ + 128 + c] = tv;
  }
  async_wait0();
  __syncthreads();
  // convert staged f32 -> bf16
  { int r = lane >> 1, c0 = (lane & 1) * 64;
#pragma unroll 4
    for (int c = 0; c < 64; ++c) A[r * DQ + c0 + c] = f2bf(S[r * DOUT + c0 + c]); }
  __syncthreads();

  v8f acc[8];
#pragma unroll
  for (int nt = 0; nt < 8; ++nt) acc[nt] = (v8f)(0.0f);
#pragma unroll
  for (int kt = 0; kt < 6; ++kt) {
    v16bf a = load_a_frag(A, DQ, kt, lane);
#pragma unroll
    for (int nt = 0; nt < 8; ++nt)
      acc[nt] = wmma_bf16(a, load_b_frag(wqb, kt, nt, lane), acc[nt]);
  }
  // fragments -> LDS (reuse S), then coalesced store
  int mrow = (lane >> 4) * 8, col0 = lane & 15;
#pragma unroll
  for (int nt = 0; nt < 8; ++nt)
#pragma unroll
    for (int rr = 0; rr < 8; ++rr) {
      int ocol = nt * 16 + col0;
      S[(rr + mrow) * DOUT + ocol] = acc[nt][rr] + wq_b[ocol];
    }
  __syncthreads();
#pragma unroll 4
  for (int r = 0; r < 16; ++r) {
    int orow = tile * 16 + r;
    if (orow < ND)
      *(float4*)(Q + (size_t)orow * DOUT + lane * 4) = *(float4*)(S + r * DOUT + lane * 4);
  }
}

// ---------------- kernel 3: per-edge K/V GEMM + attention logits ----------------
#define EWPB 2
__global__ __launch_bounds__(32 * EWPB) void k_edge(
    const float* __restrict__ h, const float* __restrict__ dt,
    const float* __restrict__ ef, const float* __restrict__ time_w,
    const float* __restrict__ time_b,
    const __bf16* __restrict__ wkb, const float* __restrict__ wk_b,
    const __bf16* __restrict__ wvb, const float* __restrict__ wv_b,
    const float* __restrict__ Q, const int* __restrict__ nbr_idx,
    const int* __restrict__ edge_dst,
    float* __restrict__ Vbuf, float* __restrict__ att,
    unsigned* __restrict__ mEnc) {
  __shared__ float  sS[EWPB][16 * 256];   // 16 KB: staging (16x192 f32) -> K(16x128)|V(16x128)
  __shared__ __bf16 sA[EWPB][16 * DK];    // 8 KB
  int wave = threadIdx.x >> 5, lane = threadIdx.x & 31;
  int tile = blockIdx.x * EWPB + wave;
  float*  S = sS[wave];
  __bf16* A = sA[wave];
  int lane16 = lane & 15;

  // async gather: h_src row (512B, 1 row/instr), staged layout S[r*192 + c]
#pragma unroll 4
  for (int r = 0; r < 16; ++r) {
    int e = tile * 16 + r;
    int ec = (e < EE) ? e : (EE - 1);
    async_ld16(h + (size_t)(ND + nbr_idx[ec]) * DNODE + lane * 4, S + r * 192 + lane * 4);
  }
  // ef rows (256B): two rows per instruction
#pragma unroll 4
  for (int q = 0; q < 8; ++q) {
    int r = q * 2 + (lane >> 4);
    int e = tile * 16 + r;
    int ec = (e < EE) ? e : (EE - 1);
    async_ld16(ef + (size_t)ec * DEDGE + lane16 * 4, S + r * 192 + 128 + lane16 * 4);
  }
  // time features straight to bf16 A (cols 192..255) while gathers fly
  { int r = lane >> 1;
    int e = tile * 16 + r;
    int ec = (e < EE) ? e : (EE - 1);
    float d = dt[ec];
    int c0 = (lane & 1) * 32;
#pragma unroll 4
    for (int c = 0; c < 32; ++c) {
      int cc = c0 + c;
      A[r * DK + 192 + cc] = f2bf(__cosf(d * time_w[cc] + time_b[cc]));
    } }
  async_wait0();
  __syncthreads();
  // convert staged f32 -> bf16 (cols 0..191)
  { int r = lane >> 1, c0 = (lane & 1) * 96;
#pragma unroll 4
    for (int c = 0; c < 96; ++c) A[r * DK + c0 + c] = f2bf(S[r * 192 + c0 + c]); }
  __syncthreads();

  v8f accK[8], accV[8];
#pragma unroll
  for (int nt = 0; nt < 8; ++nt) { accK[nt] = (v8f)(0.0f); accV[nt] = (v8f)(0.0f); }
#pragma unroll
  for (int kt = 0; kt < 8; ++kt) {
    v16bf a = load_a_frag(A, DK, kt, lane);    // shared A-fragment for both GEMMs
#pragma unroll
    for (int nt = 0; nt < 8; ++nt) {
      accK[nt] = wmma_bf16(a, load_b_frag(wkb, kt, nt, lane), accK[nt]);
      accV[nt] = wmma_bf16(a, load_b_frag(wvb, kt, nt, lane), accV[nt]);
    }
  }

  // K and V fragments -> LDS (reuse staging buffer)
  float* KK = S;            // 16 x 128
  float* VV = S + 16 * 128; // 16 x 128
  int mrow = (lane >> 4) * 8, col0 = lane & 15;
#pragma unroll
  for (int nt = 0; nt < 8; ++nt)
#pragma unroll
    for (int rr = 0; rr < 8; ++rr) {
      int lrow = rr + mrow;
      int ocol = nt * 16 + col0;
      KK[lrow * DOUT + ocol] = accK[nt][rr] + wk_b[ocol];
      VV[lrow * DOUT + ocol] = accV[nt][rr] + wv_b[ocol];
    }
  __syncthreads();

  // coalesced (async) V spill: one row (512B) per instruction
#pragma unroll 4
  for (int r = 0; r < 16; ++r) {
    int e = tile * 16 + r;
    if (e < EE)
      async_st16(Vbuf + (size_t)e * DOUT + lane * 4, VV + r * DOUT + lane * 4);
  }

  // per-(edge,head) logit: lanes 0-15 head0, 16-31 head1
  int m = lane16, hd = lane >> 4;
  int e2 = tile * 16 + m;
  if (e2 < EE) {
    int dst = edge_dst[e2];
    const float* qp = Q + (size_t)dst * DOUT + hd * HDIM;
    const float* kp = KK + m * DOUT + hd * HDIM;
    float s = 0.0f;
#pragma unroll 4
    for (int d = 0; d < HDIM; ++d) s += qp[d] * kp[d];
    s = (s > 0.0f) ? s : 0.2f * s;                 // leaky_relu(0.2)
    att[(size_t)e2 * NHEAD + hd] = s;
    atomicMax(&mEnc[dst * NHEAD + hd], fenc(s));   // segment max
  }
}

// ---------------- kernel 4: exp + segment sum of exp ----------------
__global__ void k_exp(const int* __restrict__ edge_dst, const unsigned* __restrict__ mEnc,
                      float* __restrict__ att, float* __restrict__ den) {
  int i = blockIdx.x * blockDim.x + threadIdx.x;
  if (i >= EE * NHEAD) return;
  int e = i >> 1, hd = i & 1;
  int dst = edge_dst[e];
  float mm = fdec(mEnc[dst * NHEAD + hd]);
  if (!isfinite(mm)) mm = 0.0f;
  float ex = __expf(att[i] - mm);
  att[i] = ex;
  atomicAdd(&den[dst * NHEAD + hd], ex);
}

// ---------------- kernel 5: weighted-V scatter add ----------------
__global__ void k_agg(const int* __restrict__ edge_dst, const float* __restrict__ Vbuf,
                      const float* __restrict__ att, float* __restrict__ agg) {
  size_t i = (size_t)blockIdx.x * blockDim.x + threadIdx.x;
  if (i >= (size_t)EE * DOUT) return;
  int e = (int)(i >> 7), d = (int)(i & 127);
  float w = att[(size_t)e * NHEAD + (d >> 6)];
  atomicAdd(&agg[(size_t)edge_dst[e] * DOUT + d], Vbuf[i] * w);
}

// ---------------- kernel 6: output GEMM + ReLU + LayerNorm ----------------
#define OWPB 2
__global__ __launch_bounds__(32 * OWPB) void k_out(
    const float* __restrict__ h, const float* __restrict__ agg,
    const float* __restrict__ den, const __bf16* __restrict__ wob,
    const float* __restrict__ wo_b, const float* __restrict__ ln_g,
    const float* __restrict__ ln_b, float* __restrict__ out) {
  __shared__ float  sS[OWPB][16 * 256];   // 16 KB: staging agg|h f32 -> P(16x128)
  __shared__ __bf16 sA[OWPB][16 * 256];   // 8 KB
  __shared__ float  sMu[OWPB][16], sRs[OWPB][16];
  int wave = threadIdx.x >> 5, lane = threadIdx.x & 31;
  int tile = blockIdx.x * OWPB + wave;
  float*  S = sS[wave];
  __bf16* A = sA[wave];

#pragma unroll 4
  for (int r = 0; r < 16; ++r) {
    int rr = tile * 16 + r;
    int rc = (rr < ND) ? rr : (ND - 1);
    async_ld16(agg + (size_t)rc * DOUT + lane * 4, S + r * 256 + lane * 4);
  }
#pragma unroll 4
  for (int r = 0; r < 16; ++r) {
    int rr = tile * 16 + r;
    int rc = (rr < ND) ? rr : (ND - 1);
    async_ld16(h + (size_t)rc * DNODE + lane * 4, S + r * 256 + 128 + lane * 4);
  }
  async_wait0();
  __syncthreads();

  // convert: softmax normalization on agg half, plain convert on h half
  { int r = lane >> 1;
    if (lane & 1) {
#pragma unroll 4
      for (int c = 0; c < 128; ++c) A[r * 256 + 128 + c] = f2bf(S[r * 256 + 128 + c]);
    } else {
      int rr = tile * 16 + r;
      int rc = (rr < ND) ? rr : (ND - 1);
      float d0 = den[rc * 2], d1 = den[rc * 2 + 1];
      float i0 = (d0 > 0.0f) ? 1.0f / d0 : 0.0f;
      float i1 = (d1 > 0.0f) ? 1.0f / d1 : 0.0f;
#pragma unroll 4
      for (int c = 0; c < 64; ++c)   A[r * 256 + c]      = f2bf(S[r * 256 + c] * i0);
#pragma unroll 4
      for (int c = 64; c < 128; ++c) A[r * 256 + c]      = f2bf(S[r * 256 + c] * i1);
    } }
  __syncthreads();

  v8f acc[8];
#pragma unroll
  for (int nt = 0; nt < 8; ++nt) acc[nt] = (v8f)(0.0f);
#pragma unroll
  for (int kt = 0; kt < 8; ++kt) {
    v16bf a = load_a_frag(A, 256, kt, lane);
#pragma unroll
    for (int nt = 0; nt < 8; ++nt)
      acc[nt] = wmma_bf16(a, load_b_frag(wob, kt, nt, lane), acc[nt]);
  }
  // bias + ReLU -> P (reuse S[0..2048))
  int mrow = (lane >> 4) * 8, col0 = lane & 15;
#pragma unroll
  for (int nt = 0; nt < 8; ++nt)
#pragma unroll
    for (int rr = 0; rr < 8; ++rr) {
      int ocol = nt * 16 + col0;
      float v = acc[nt][rr] + wo_b[ocol];
      S[(rr + mrow) * DOUT + ocol] = (v > 0.0f) ? v : 0.0f;
    }
  __syncthreads();

  // per-row layernorm stats
  if (lane < 16) {
    const float* pr = S + lane * DOUT;
    float s = 0.0f, ss = 0.0f;
#pragma unroll 4
    for (int c = 0; c < DOUT; ++c) { float x = pr[c]; s += x; ss += x * x; }
    float mu = s * (1.0f / DOUT);
    float var = ss * (1.0f / DOUT) - mu * mu;
    sMu[wave][lane] = mu;
    sRs[wave][lane] = rsqrtf(var + LN_EPS);
  }
  __syncthreads();

  // coalesced normalized store: one row (512B) per iteration
#pragma unroll 4
  for (int r = 0; r < 16; ++r) {
    int orow = tile * 16 + r;
    if (orow >= ND) continue;
    float mu = sMu[wave][r], rs = sRs[wave][r];
    int c = lane * 4;
    float4 x = *(float4*)(S + r * DOUT + c);
    float4 g = *(const float4*)(ln_g + c);
    float4 b = *(const float4*)(ln_b + c);
    float4 o;
    o.x = (x.x - mu) * rs * g.x + b.x;
    o.y = (x.y - mu) * rs * g.y + b.y;
    o.z = (x.z - mu) * rs * g.z + b.z;
    o.w = (x.w - mu) * rs * g.w + b.w;
    *(float4*)(out + (size_t)orow * DOUT + c) = o;
  }
}

// ---------------- host launcher ----------------
extern "C" void kernel_launch(void* const* d_in, const int* in_sizes, int n_in,
                              void* d_out, int out_size, void* d_ws, size_t ws_size,
                              hipStream_t stream) {
  const float* h      = (const float*)d_in[0];
  const float* dt     = (const float*)d_in[1];
  const float* ef     = (const float*)d_in[2];
  const float* time_w = (const float*)d_in[3];
  const float* time_b = (const float*)d_in[4];
  const float* wq_w   = (const float*)d_in[5];
  const float* wq_b   = (const float*)d_in[6];
  const float* wk_w   = (const float*)d_in[7];
  const float* wk_b   = (const float*)d_in[8];
  const float* wv_w   = (const float*)d_in[9];
  const float* wv_b   = (const float*)d_in[10];
  const float* wo_w   = (const float*)d_in[11];
  const float* wo_b   = (const float*)d_in[12];
  const float* ln_g   = (const float*)d_in[13];
  const float* ln_b   = (const float*)d_in[14];
  const int* nbr_idx  = (const int*)d_in[15];
  const int* edge_dst = (const int*)d_in[16];
  float* out = (float*)d_out;

  // workspace carve-up (all chunks 32B aligned)
  float* Q    = (float*)d_ws;                               // ND*128
  float* Vbuf = Q + (size_t)ND * DOUT;                      // E*128
  float* att  = Vbuf + (size_t)EE * DOUT;                   // E*2
  unsigned* mEnc = (unsigned*)(att + (size_t)EE * NHEAD);   // ND*2
  float* den  = (float*)(mEnc + (size_t)ND * NHEAD);        // ND*2
  float* agg  = den + (size_t)ND * NHEAD;                   // ND*128
  __bf16* wqb = (__bf16*)(agg + (size_t)ND * DOUT);         // 192*128
  __bf16* wkb = wqb + 192 * 128;                            // 256*128
  __bf16* wvb = wkb + 256 * 128;
  __bf16* wob = wvb + 256 * 128;

  { int n = ND * DOUT; k_init<<<(n + 255) / 256, 256, 0, stream>>>(agg, den, mEnc); }
  k_prep<<<(192 * 128 + 255) / 256, 256, 0, stream>>>(wq_w, wqb, 192);
  k_prep<<<(256 * 128 + 255) / 256, 256, 0, stream>>>(wk_w, wkb, 256);
  k_prep<<<(256 * 128 + 255) / 256, 256, 0, stream>>>(wv_w, wvb, 256);
  k_prep<<<(256 * 128 + 255) / 256, 256, 0, stream>>>(wo_w, wob, 256);
  { int tiles = (ND + 15) / 16;
    k_q<<<(tiles + QWPB - 1) / QWPB, 32 * QWPB, 0, stream>>>(h, time_b, wqb, wq_b, Q); }
  { int tiles = (EE + 15) / 16;
    k_edge<<<(tiles + EWPB - 1) / EWPB, 32 * EWPB, 0, stream>>>(
        h, dt, ef, time_w, time_b, wkb, wk_b, wvb, wv_b, Q,
        nbr_idx, edge_dst, Vbuf, att, mEnc); }
  { int n = EE * NHEAD; k_exp<<<(n + 255) / 256, 256, 0, stream>>>(edge_dst, mEnc, att, den); }
  { size_t n = (size_t)EE * DOUT;
    k_agg<<<(unsigned)((n + 255) / 256), 256, 0, stream>>>(edge_dst, Vbuf, att, agg); }
  { int tiles = (ND + 15) / 16;
    k_out<<<(tiles + OWPB - 1) / OWPB, 32 * OWPB, 0, stream>>>(
        h, agg, den, wob, wo_b, ln_g, ln_b, out); }
}
